// MetaLayer_11003706212370
// MI455X (gfx1250) — matrix operand order repaired and checked by reading, not compiled, not run on hardware
//
#include <hip/hip_runtime.h>

#define AP  132   // A-tile pitch (u32 kpairs): lane stride ≡ 4 banks -> conflict-free b128 rows
#define ENP 36    // edge_new tile pitch: 16B-aligned stride, conflict-free b128 rows

typedef __bf16 bf16_t;
typedef bf16_t v16bf __attribute__((ext_vector_type(16)));
typedef float  v8f   __attribute__((ext_vector_type(8)));

union AB16 { v16bf v; unsigned u[8]; };

struct Src { const unsigned* p; int pitch; };

// ---------------- helpers ----------------
__device__ __forceinline__ unsigned short f2bf(float f) {
  unsigned u = __float_as_uint(f);
  u += 0x7FFFu + ((u >> 16) & 1u);          // round-to-nearest-even
  return (unsigned short)(u >> 16);
}
__device__ __forceinline__ unsigned pack2bf(float a, float b) {
  return (unsigned)f2bf(a) | ((unsigned)f2bf(b) << 16);
}
__device__ __forceinline__ float bfbits2f(unsigned h) { return __uint_as_float(h << 16); }

// monotone float <-> uint encoding for atomicMax on floats (0 == -inf identity)
__device__ __forceinline__ unsigned fenc(float f) {
  unsigned i = __float_as_uint(f);
  return (i & 0x80000000u) ? ~i : (i | 0x80000000u);
}
__device__ __forceinline__ float fdec(unsigned u) {
  return __uint_as_float((u & 0x80000000u) ? (u & 0x7FFFFFFFu) : ~u);
}

// A fragment (16x32 bf16) per ISA 7.12.2: lanes 0-15 row m, VGPR j<4 -> Kpair kp0+half*4+j,
// VGPR j>=4 -> Kpair kp0+8+half*4+(j-4).  Two 16B-aligned contiguous quads -> 2x ds_load_b128.
__device__ __forceinline__ void loadA(const unsigned* base, int pitch, int rbase, int kp0, AB16& A) {
  int lane = threadIdx.x & 31;
  const unsigned* rp = base + (size_t)(rbase + (lane & 15)) * pitch + kp0 + (lane >> 4) * 4;
  *(uint4*)&A.u[0] = *(const uint4*)rp;
  *(uint4*)&A.u[4] = *(const uint4*)(rp + 8);
}

// B fragment, fragment-major staged layout: frag*256 + lane*4 (+128 for upper quad).
// 2x ds_load_b128, conflict-free, results land contiguous (no register shuffles).
__device__ __forceinline__ void loadBf(const unsigned* base, int frag, AB16& B) {
  int lane = threadIdx.x & 31;
  const unsigned* p = base + frag * 256 + lane * 4;
  *(uint4*)&B.u[0] = *(const uint4*)p;
  *(uint4*)&B.u[4] = *(const uint4*)(p + 128);
}

__device__ __forceinline__ v8f bmm(const AB16& a, const AB16& b, v8f c) {
  return __builtin_amdgcn_wmma_f32_16x16x32_bf16(false, a.v, false, b.v, (short)0, c, false, false);
}

// stage weight matrix W [64 x K] (torch [out,in]) into fragment-major bf16-pair LDS.
// frag f = ks*4+nt holds B-fragment for k-step ks, n-tile nt.
// element (lane, j): kpair = ks*16 + (j>>2)*8 + (lane>>4)*4 + (j&3) ; n = nt*16 + (lane&15)
// dst linear index == i by construction.
__device__ __forceinline__ void stageW(int tid, const float* __restrict__ W, int K, int nsteps,
                                       unsigned* dst) {
  int total = nsteps * 4 * 256;
  for (int i = tid; i < total; i += 256) {
    int f = i >> 8, rem = i & 255;
    int jh = rem >> 7, lane = (rem >> 2) & 31, jj = rem & 3;
    int j = jh * 4 + jj;
    int ks = f >> 2, nt = f & 3;
    int kpair = ks * 16 + jh * 8 + (lane >> 4) * 4 + jj;
    int n = nt * 16 + (lane & 15);
    float2 w = *(const float2*)(W + (size_t)n * K + 2 * kpair);
    dst[i] = pack2bf(w.x, w.y);
  }
}

__device__ __forceinline__ void gemm_steps(const Src* st, int nsteps, const unsigned* Bm,
                                           int rbase, v8f acc[4]) {
  for (int ks = 0; ks < nsteps; ++ks) {
    AB16 a; loadA(st[ks].p, st[ks].pitch, rbase, 0, a);
#pragma unroll
    for (int nt = 0; nt < 4; ++nt) {
      AB16 b; loadBf(Bm, ks * 4 + nt, b);
      acc[nt] = bmm(a, b, acc[nt]);
    }
  }
}

// h = leaky_relu(acc + b1); w = sum_n h*w2 -> per-row scalar via butterfly over 16 lanes
__device__ __forceinline__ void attn_w(v8f acc[4], const float b1l[4], const float w2l[4],
                                       int rbase, long tile0, int E, float* out) {
  int lane = threadIdx.x & 31, half = lane >> 4, ml = lane & 15;
  float wp[8] = {0, 0, 0, 0, 0, 0, 0, 0};
#pragma unroll
  for (int nt = 0; nt < 4; ++nt)
#pragma unroll
    for (int r = 0; r < 8; ++r) {
      float hv = acc[nt][r] + b1l[nt];
      hv = hv > 0.f ? hv : 0.01f * hv;
      wp[r] += hv * w2l[nt];
    }
#pragma unroll
  for (int r = 0; r < 8; ++r) {
    float s = wp[r];
    s += __shfl_xor(s, 1, 32); s += __shfl_xor(s, 2, 32);
    s += __shfl_xor(s, 4, 32); s += __shfl_xor(s, 8, 32);
    if (ml == 0) {
      long e = tile0 + rbase + half * 8 + r;
      if (e < E) out[e] = s;
    }
  }
}

// v = acc + b3, stored as packed bf16 pairs [E][32] u32
__device__ __forceinline__ void attn_v(v8f acc[4], const float b3l[4],
                                       int rbase, long tile0, int E, unsigned* out) {
  int lane = threadIdx.x & 31, half = lane >> 4, ml = lane & 15;
#pragma unroll
  for (int nt = 0; nt < 4; ++nt)
#pragma unroll
    for (int r = 0; r < 8; ++r) {
      float vv = acc[nt][r] + b3l[nt];
      float ov = __shfl_xor(vv, 1, 32);
      if (!(ml & 1)) {
        long e = tile0 + rbase + half * 8 + r;
        if (e < E) out[e * 32 + nt * 8 + (ml >> 1)] = pack2bf(vv, ov);
      }
    }
}

__device__ __forceinline__ int lowerBound(const int* a, int n, int v) {
  int lo = 0, hi = n;
  while (lo < hi) { int mid = (lo + hi) >> 1; if (a[mid] < v) lo = mid + 1; else hi = mid; }
  return lo;
}

// =============== kernel 1: fused edge MLP + attention logits/values ===============
__global__ __launch_bounds__(256) void edge_attn_kernel(
    const float* __restrict__ x, const float* __restrict__ ea, const float* __restrict__ u,
    const float* __restrict__ W_edge, const float* __restrict__ b_edge,
    const float* __restrict__ na_w1, const float* __restrict__ na_b1,
    const float* __restrict__ na_w2, const float* __restrict__ na_w3,
    const float* __restrict__ na_b3,
    const int* __restrict__ rowG, const int* __restrict__ colG, const int* __restrict__ ebG,
    float* __restrict__ edge_new, float* __restrict__ wS, float* __restrict__ wR,
    unsigned* __restrict__ vS, unsigned* __restrict__ vR, int E)
{
  __shared__ unsigned Ap[128 * AP];     // [e_attr | x[row] | x[col] | u[eb]] bf16 pairs
  __shared__ unsigned ENp[128 * ENP];   // edge_new bf16 pairs
  __shared__ unsigned WEf[32 * 256];    // W_edge^T fragments (8 ksteps x 4 ntiles)
  __shared__ unsigned W1f[24 * 256];    // na_w1^T fragments (6 x 4)
  __shared__ unsigned W3f[16 * 256];    // na_w3^T fragments (4 x 4)
  __shared__ float w2s[64], bEs[64], b1s[64], b3s[64];
  __shared__ int rI[128], cI[128], bI[128];

  const int tid = threadIdx.x;
  const long tile0 = (long)blockIdx.x * 128;

  // stage weights in fragment-major bf16 layout
  stageW(tid, W_edge, 256, 8, WEf);
  stageW(tid, na_w1, 192, 6, W1f);
  stageW(tid, na_w3, 128, 4, W3f);
  if (tid < 64) { w2s[tid] = na_w2[tid]; bEs[tid] = b_edge[tid]; b1s[tid] = na_b1[tid]; b3s[tid] = na_b3[tid]; }
  if (tid < 128) {
    long e = tile0 + tid; if (e > (long)E - 1) e = E - 1;
    rI[tid] = rowG[e]; cI[tid] = colG[e]; bI[tid] = ebG[e];
  }
  __syncthreads();

  // gather + pack A tile: 128 edges x 128 kpairs
  for (int q = tid; q < 128 * 128; q += 256) {
    int er = q >> 7, p = q & 127;
    long e = tile0 + er; if (e > (long)E - 1) e = E - 1;
    const float* s;
    if (p < 32)      s = ea + e * 64 + 2 * p;
    else if (p < 64) s = x + (long)rI[er] * 64 + 2 * (p - 32);
    else if (p < 96) s = x + (long)cI[er] * 64 + 2 * (p - 64);
    else             s = u + (long)bI[er] * 64 + 2 * (p - 96);
    float2 f = *(const float2*)s;
    Ap[er * AP + p] = pack2bf(f.x, f.y);
  }
  __syncthreads();

  const int lane = tid & 31, half = lane >> 4, ml = lane & 15;
  const int rbase = (tid >> 5) * 16;
  const v8f z = {0, 0, 0, 0, 0, 0, 0, 0};

  // ---- edge GEMM [16x256]x[256x64] ----
  v8f accE[4] = {z, z, z, z};
  for (int ks = 0; ks < 8; ++ks) {
    AB16 a; loadA(Ap, AP, rbase, ks * 16, a);
#pragma unroll
    for (int nt = 0; nt < 4; ++nt) {
      AB16 b; loadBf(WEf, ks * 4 + nt, b);
      accE[nt] = bmm(a, b, accE[nt]);
    }
  }
  unsigned short* ENh = (unsigned short*)ENp;
#pragma unroll
  for (int nt = 0; nt < 4; ++nt) {
    int n = nt * 16 + ml;
    float bb = bEs[n];
#pragma unroll
    for (int r = 0; r < 8; ++r) {
      float vv = accE[nt][r] + bb;
      vv = vv > 0.f ? vv : 0.f;
      int rl = rbase + half * 8 + r;
      ENh[rl * (ENP * 2) + n] = f2bf(vv);      // keep for attention A-operands (own rows only)
      long e = tile0 + rl;
      if (e < E) edge_new[e * 64 + n] = vv;
    }
  }

  float w2l[4], b1l[4], b3l[4];
#pragma unroll
  for (int nt = 0; nt < 4; ++nt) {
    w2l[nt] = w2s[nt * 16 + ml]; b1l[nt] = b1s[nt * 16 + ml]; b3l[nt] = b3s[nt * 16 + ml];
  }

  // ---- h_sent = [x[row]|x[col]|edge_new] @ na_w1^T ----
  {
    Src st[6] = {{Ap + 32, AP}, {Ap + 48, AP}, {Ap + 64, AP}, {Ap + 80, AP}, {ENp, ENP}, {ENp + 16, ENP}};
    v8f acc[4] = {z, z, z, z};
    gemm_steps(st, 6, W1f, rbase, acc);
    attn_w(acc, b1l, w2l, rbase, tile0, E, wS);
  }
  // ---- h_recv = [x[col]|x[row]|edge_new] @ na_w1^T ----
  {
    Src st[6] = {{Ap + 64, AP}, {Ap + 80, AP}, {Ap + 32, AP}, {Ap + 48, AP}, {ENp, ENP}, {ENp + 16, ENP}};
    v8f acc[4] = {z, z, z, z};
    gemm_steps(st, 6, W1f, rbase, acc);
    attn_w(acc, b1l, w2l, rbase, tile0, E, wR);
  }
  // ---- v_sent = [x[col]|edge_new] @ na_w3^T ----
  {
    Src st[4] = {{Ap + 64, AP}, {Ap + 80, AP}, {ENp, ENP}, {ENp + 16, ENP}};
    v8f acc[4] = {z, z, z, z};
    gemm_steps(st, 4, W3f, rbase, acc);
    attn_v(acc, b3l, rbase, tile0, E, vS);
  }
  // ---- v_recv = [x[row]|edge_new] @ na_w3^T ----
  {
    Src st[4] = {{Ap + 32, AP}, {Ap + 48, AP}, {ENp, ENP}, {ENp + 16, ENP}};
    v8f acc[4] = {z, z, z, z};
    gemm_steps(st, 4, W3f, rbase, acc);
    attn_v(acc, b3l, rbase, tile0, E, vR);
  }
}

// =============== kernel: zero scratch ===============
__global__ void zero_kernel(float* p, long n) {
  long i = (long)blockIdx.x * blockDim.x + threadIdx.x;
  long stride = (long)gridDim.x * blockDim.x;
  for (; i < n; i += stride) p[i] = 0.f;
}

// =============== kernel: segment max (ordered-uint atomicMax) ===============
__global__ void attn_max_kernel(const float* __restrict__ wS, const float* __restrict__ wR,
                                const int* __restrict__ rowG, const int* __restrict__ colG,
                                unsigned* __restrict__ mS, unsigned* __restrict__ mR, int E) {
  int e = blockIdx.x * 256 + threadIdx.x;
  if (e < E) {
    atomicMax(&mS[rowG[e]], fenc(wS[e]));
    atomicMax(&mR[colG[e]], fenc(wR[e]));
  }
}

// =============== kernel: exp(w - m), segment sum ===============
__global__ void attn_expsum_kernel(float* __restrict__ wS, float* __restrict__ wR,
                                   const int* __restrict__ rowG, const int* __restrict__ colG,
                                   const unsigned* __restrict__ mS, const unsigned* __restrict__ mR,
                                   float* __restrict__ smS, float* __restrict__ smR, int E) {
  int e = blockIdx.x * 256 + threadIdx.x;
  if (e < E) {
    int r = rowG[e], c = colG[e];
    float es = expf(wS[e] - fdec(mS[r]));
    float er = expf(wR[e] - fdec(mR[c]));
    wS[e] = es; wR[e] = er;                 // overwrite logits with exp values
    atomicAdd(&smS[r], es);
    atomicAdd(&smR[c], er);
  }
}

// =============== kernel: scatter p*v into sent/recv ===============
__global__ void attn_scatter_kernel(const float* __restrict__ exS, const float* __restrict__ exR,
                                    const int* __restrict__ rowG, const int* __restrict__ colG,
                                    const float* __restrict__ smS, const float* __restrict__ smR,
                                    const unsigned* __restrict__ vS, const unsigned* __restrict__ vR,
                                    float* __restrict__ sent, float* __restrict__ recv, int E) {
  long idx = (long)blockIdx.x * 256 + threadIdx.x;
  if (idx >= (long)E * 32) return;
  int e = (int)(idx >> 5), pr = (int)(idx & 31);
  int r = rowG[e], c = colG[e];
  float ps = exS[e] / (smS[r] + 1e-16f);
  float pc = exR[e] / (smR[c] + 1e-16f);
  unsigned a = vS[(long)e * 32 + pr];
  unsigned b = vR[(long)e * 32 + pr];
  atomicAdd(&sent[(long)r * 64 + 2 * pr],     ps * bfbits2f(a & 0xFFFFu));
  atomicAdd(&sent[(long)r * 64 + 2 * pr + 1], ps * bfbits2f(a >> 16));
  atomicAdd(&recv[(long)c * 64 + 2 * pr],     pc * bfbits2f(b & 0xFFFFu));
  atomicAdd(&recv[(long)c * 64 + 2 * pr + 1], pc * bfbits2f(b >> 16));
}

// =============== kernel: node MLP [N x 256] x [256 x 64] ===============
__global__ __launch_bounds__(256) void node_kernel(
    const float* __restrict__ x, const float* __restrict__ sent, const float* __restrict__ recv,
    const float* __restrict__ u, const float* __restrict__ W_node, const float* __restrict__ b_node,
    const int* __restrict__ nb, float* __restrict__ x_new, int N)
{
  __shared__ unsigned Ap[128 * AP];
  __shared__ unsigned WNf[32 * 256];
  __shared__ float bNs[64];
  __shared__ int gI[128];
  const int tid = threadIdx.x;
  const long tile0 = (long)blockIdx.x * 128;

  stageW(tid, W_node, 256, 8, WNf);
  if (tid < 64) bNs[tid] = b_node[tid];
  if (tid < 128) { long e = tile0 + tid; if (e > (long)N - 1) e = N - 1; gI[tid] = nb[e]; }
  __syncthreads();

  for (int q = tid; q < 128 * 128; q += 256) {
    int er = q >> 7, p = q & 127;
    long e = tile0 + er; if (e > (long)N - 1) e = N - 1;
    const float* s;
    if (p < 32)      s = x + e * 64 + 2 * p;
    else if (p < 64) s = sent + e * 64 + 2 * (p - 32);
    else if (p < 96) s = recv + e * 64 + 2 * (p - 64);
    else             s = u + (long)gI[er] * 64 + 2 * (p - 96);
    float2 f = *(const float2*)s;
    Ap[er * AP + p] = pack2bf(f.x, f.y);
  }
  __syncthreads();

  const int lane = tid & 31, half = lane >> 4, ml = lane & 15;
  const int rbase = (tid >> 5) * 16;
  const v8f z = {0, 0, 0, 0, 0, 0, 0, 0};
  v8f acc[4] = {z, z, z, z};
  for (int ks = 0; ks < 8; ++ks) {
    AB16 a; loadA(Ap, AP, rbase, ks * 16, a);
#pragma unroll
    for (int nt = 0; nt < 4; ++nt) {
      AB16 b; loadBf(WNf, ks * 4 + nt, b);
      acc[nt] = bmm(a, b, acc[nt]);
    }
  }
#pragma unroll
  for (int nt = 0; nt < 4; ++nt) {
    int n = nt * 16 + ml;
    float bb = bNs[n];
#pragma unroll
    for (int r = 0; r < 8; ++r) {
      float vv = acc[nt][r] + bb;
      vv = vv > 0.f ? vv : 0.f;
      long e = tile0 + rbase + half * 8 + r;
      if (e < N) x_new[e * 64 + n] = vv;
    }
  }
}

// =============== kernel: per-graph aggregation (sorted batches, binary search) ===============
__global__ __launch_bounds__(256) void agg_kernel(
    const float* __restrict__ xn, const int* __restrict__ nb, int N,
    const float* __restrict__ en, const int* __restrict__ ebat, int E,
    float* __restrict__ nagg, float* __restrict__ eagg, int G)
{
  const int CH = 8;
  int g = blockIdx.x / CH, c = blockIdx.x % CH;
  const float* src; const int* bat; int len; float* out;
  if (blockIdx.y == 0) { src = xn; bat = nb;   len = N; out = nagg; }
  else                 { src = en; bat = ebat; len = E; out = eagg; }
  int lo = lowerBound(bat, len, g), hi = lowerBound(bat, len, g + 1);
  int cnt = hi - lo;
  int s = lo + (int)((long)cnt * c / CH);
  int t = lo + (int)((long)cnt * (c + 1) / CH);
  int colp = threadIdx.x & 63, rr = threadIdx.x >> 6;
  float a = 0.f;
  for (int i = s + rr; i < t; i += 4) a += src[(long)i * 64 + colp];
  __shared__ float red[4 * 64];
  red[rr * 64 + colp] = a;
  __syncthreads();
  if (threadIdx.x < 64) {
    float v = red[threadIdx.x] + red[64 + threadIdx.x] + red[128 + threadIdx.x] + red[192 + threadIdx.x];
    atomicAdd(out + (long)g * 64 + threadIdx.x, v);
  }
}

// =============== kernel: global MLP [G x 192] x [192 x 64] (tiny) ===============
__global__ void glob_kernel(const float* __restrict__ u, const float* __restrict__ nagg,
                            const float* __restrict__ eagg, const float* __restrict__ W,
                            const float* __restrict__ b, float* __restrict__ out) {
  int g = blockIdx.x, n = threadIdx.x;
  const float* wr = W + (size_t)n * 192;
  float acc = b[n];
  for (int k = 0; k < 64; ++k) acc += u[g * 64 + k] * wr[k];
  for (int k = 0; k < 64; ++k) acc += nagg[g * 64 + k] * wr[64 + k];
  for (int k = 0; k < 64; ++k) acc += eagg[g * 64 + k] * wr[128 + k];
  out[g * 64 + n] = acc > 0.f ? acc : 0.f;
}

extern "C" void kernel_launch(void* const* d_in, const int* in_sizes, int n_in,
                              void* d_out, int out_size, void* d_ws, size_t ws_size,
                              hipStream_t stream) {
  const float* x      = (const float*)d_in[0];
  const float* ea     = (const float*)d_in[1];
  const float* u      = (const float*)d_in[2];
  const float* W_edge = (const float*)d_in[3];
  const float* b_edge = (const float*)d_in[4];
  const float* W_node = (const float*)d_in[5];
  const float* b_node = (const float*)d_in[6];
  const float* W_glob = (const float*)d_in[7];
  const float* b_glob = (const float*)d_in[8];
  const float* na_w1  = (const float*)d_in[9];
  const float* na_b1  = (const float*)d_in[10];
  const float* na_w2  = (const float*)d_in[11];
  const float* na_w3  = (const float*)d_in[12];
  const float* na_b3  = (const float*)d_in[13];
  const int* eidx     = (const int*)d_in[14];
  const int* node_b   = (const int*)d_in[15];
  const int* edge_b   = (const int*)d_in[16];

  const int N = in_sizes[0] / 64;
  const int E = in_sizes[1] / 64;
  const int G = in_sizes[2] / 64;

  const int* rowG = eidx;
  const int* colG = eidx + E;

  float* x_new = (float*)d_out;                  // [N*64]
  float* e_new = x_new + (long)N * 64;           // [E*64]
  float* u_new = e_new + (long)E * 64;           // [G*64]

  // workspace layout (f32 units)
  float* wsf = (float*)d_ws;
  float*    wS  = wsf;                           // [E]  logits -> exp
  float*    wR  = wS + E;                        // [E]
  unsigned* vS  = (unsigned*)(wR + E);           // [E*32] bf16 pairs
  unsigned* vR  = vS + (long)E * 32;             // [E*32]
  unsigned* mS  = vR + (long)E * 32;             // [N] ordered-uint max
  unsigned* mR  = mS + N;                        // [N]
  float*    smS = (float*)(mR + N);              // [N]
  float*    smR = smS + N;                       // [N]
  float*    sent = smR + N;                      // [N*64]
  float*    recv = sent + (long)N * 64;          // [N*64]
  float*    nagg = recv + (long)N * 64;          // [G*64]
  float*    eagg = nagg + (long)G * 64;          // [G*64]

  // 1) zero m/sm/sent/recv/aggs (enc(-inf) identity == 0)
  long zcount = (long)N * 132 + (long)G * 128;
  zero_kernel<<<2048, 256, 0, stream>>>((float*)mS, zcount);

  // 2) fused edge MLP + attention logits/values
  edge_attn_kernel<<<(E + 127) / 128, 256, 0, stream>>>(
      x, ea, u, W_edge, b_edge, na_w1, na_b1, na_w2, na_w3, na_b3,
      rowG, colG, edge_b, e_new, wS, wR, vS, vR, E);

  // 3) segment max
  attn_max_kernel<<<(E + 255) / 256, 256, 0, stream>>>(wS, wR, rowG, colG, mS, mR, E);

  // 4) exp + segment sum
  attn_expsum_kernel<<<(E + 255) / 256, 256, 0, stream>>>(wS, wR, rowG, colG, mS, mR, smS, smR, E);

  // 5) scatter p*v
  long sc = (long)E * 32;
  attn_scatter_kernel<<<(unsigned)((sc + 255) / 256), 256, 0, stream>>>(
      wS, wR, rowG, colG, smS, smR, vS, vR, sent, recv, E);

  // 6) node MLP
  node_kernel<<<(N + 127) / 128, 256, 0, stream>>>(
      x, sent, recv, u, W_node, b_node, node_b, x_new, N);

  // 7) per-graph aggregation (sorted batches)
  agg_kernel<<<dim3(G * 8, 2), 256, 0, stream>>>(x_new, node_b, N, e_new, edge_b, E, nagg, eagg, G);

  // 8) global MLP
  glob_kernel<<<G, 64, 0, stream>>>(u, nagg, eagg, W_glob, b_glob, u_new);
}
